// Correlation_Layer_3435973837203
// MI455X (gfx1250) — matrix operand order repaired and checked by reading, not compiled, not run on hardware
//
#include <hip/hip_runtime.h>

// Correlation layer for MI455X (gfx1250, wave32, WMMA).
// B=8, H=W=64 (HW=4096), C=256.
// out[b,q,k] = (<fB[b,q,:],fA[b,k,:]> * Wt[q,k]) / (sum_q(<..>*Wt) + 1e-8)

typedef __attribute__((ext_vector_type(16))) _Float16 v16h;
typedef __attribute__((ext_vector_type(8)))  float    v8f;

#define HW   4096
#define CCH  256
#define NB   8

// ---- fragment loaders: convert f32 rows -> f16 WMMA fragments -------------

__device__ __forceinline__ v16h cvt16(float4 p0, float4 p1, float4 p2, float4 p3) {
    v16h r;
    r[0]  = (_Float16)p0.x; r[1]  = (_Float16)p0.y; r[2]  = (_Float16)p0.z; r[3]  = (_Float16)p0.w;
    r[4]  = (_Float16)p1.x; r[5]  = (_Float16)p1.y; r[6]  = (_Float16)p1.z; r[7]  = (_Float16)p1.w;
    r[8]  = (_Float16)p2.x; r[9]  = (_Float16)p2.y; r[10] = (_Float16)p2.z; r[11] = (_Float16)p2.w;
    r[12] = (_Float16)p3.x; r[13] = (_Float16)p3.y; r[14] = (_Float16)p3.z; r[15] = (_Float16)p3.w;
    return r;
}

// A-matrix 16x32 f16 layout used here (contiguous k, same as B): lane m=l&15 -> row M=m.
// half = l>>4:  half 0 holds K 0..15, half 1 holds K 16..31 (consecutive).
// Caller pre-offsets the row pointer by (kc + half*16).
__device__ __forceinline__ v16h load_frag_A(const float* __restrict__ p) {
    return cvt16(*(const float4*)(p),      *(const float4*)(p + 4),
                 *(const float4*)(p + 8),  *(const float4*)(p + 12));
}

// B-matrix 32x16 f16 layout used here (contiguous k, same as A): lane n=l&15 -> col N=n,
// lanes 0-15 hold K=0..15, lanes 16-31 hold K=16..31 (consecutive).
// Caller pre-offsets the row pointer by (kc + half*16).
__device__ __forceinline__ v16h load_frag_B(const float* __restrict__ p) {
    return cvt16(*(const float4*)(p),     *(const float4*)(p + 4),
                 *(const float4*)(p + 8), *(const float4*)(p + 12));
}

// C/D layout: lanes 0-15: N=lane, VGPR r -> M=r ; lanes 16-31: N=lane-16, M=r+8.
__device__ __forceinline__ void store_masked_tile(v8f acc, const float* __restrict__ Wt,
                                                  float* __restrict__ out,
                                                  int b, int qbase, int half, int k) {
#pragma unroll
    for (int r = 0; r < 8; ++r) {
        int q = qbase + r + 8 * half;
        float w = Wt[(size_t)q * HW + k];
        out[((size_t)b * HW + q) * HW + k] = acc[r] * w;
    }
}

// ---- kernel 1: batched GEMM (f16 WMMA, f32 acc) + epipolar mask -----------
// block = 256 threads (8 waves). Wave w computes rows [qbase+16w, +16) x 64 k.
// grid = (HW/64, HW/128, NB)
__global__ __launch_bounds__(256)
void corr_gemm_mask(const float* __restrict__ fA, const float* __restrict__ fB,
                    const float* __restrict__ Wt, float* __restrict__ out) {
    const int lane = threadIdx.x & 31;
    const int wave = threadIdx.x >> 5;
    const int half = lane >> 4;
    const int m    = lane & 15;
    const int b    = blockIdx.z;
    const int qbase = blockIdx.y * 128 + wave * 16;
    const int kbase = blockIdx.x * 64;

    // per-lane row pointers (layout-adjusted)
    const float* fBrow = fB + ((size_t)b * HW + qbase + m) * CCH + half * 16;
    const float* fArow = fA + ((size_t)b * HW + kbase + m) * CCH + half * 16;

    v8f acc0 = {}, acc1 = {}, acc2 = {}, acc3 = {};

#pragma unroll
    for (int kc = 0; kc < CCH; kc += 32) {
        v16h a  = load_frag_A(fBrow + kc);
        v16h b0 = load_frag_B(fArow + kc);
        v16h b1 = load_frag_B(fArow + 16 * CCH + kc);
        v16h b2 = load_frag_B(fArow + 32 * CCH + kc);
        v16h b3 = load_frag_B(fArow + 48 * CCH + kc);
        acc0 = __builtin_amdgcn_wmma_f32_16x16x32_f16(false, a, false, b0, (short)0, acc0, false, false);
        acc1 = __builtin_amdgcn_wmma_f32_16x16x32_f16(false, a, false, b1, (short)0, acc1, false, false);
        acc2 = __builtin_amdgcn_wmma_f32_16x16x32_f16(false, a, false, b2, (short)0, acc2, false, false);
        acc3 = __builtin_amdgcn_wmma_f32_16x16x32_f16(false, a, false, b3, (short)0, acc3, false, false);
    }

    const int n = m;
    store_masked_tile(acc0, Wt, out, b, qbase, half, kbase + n);
    store_masked_tile(acc1, Wt, out, b, qbase, half, kbase + 16 + n);
    store_masked_tile(acc2, Wt, out, b, qbase, half, kbase + 32 + n);
    store_masked_tile(acc3, Wt, out, b, qbase, half, kbase + 48 + n);
}

// ---- kernel 2: deterministic column partial sums over q -------------------
// grid = (HW/256, 8 q-chunks, NB); each thread sums 512 q's for one k column.
__global__ __launch_bounds__(256)
void colsum_partial(const float* __restrict__ out, float* __restrict__ partial) {
    const int b  = blockIdx.z;
    const int qc = blockIdx.y;            // 0..7, chunk of 512 q
    const int k  = blockIdx.x * 256 + threadIdx.x;
    const float* base = out + ((size_t)b * HW + qc * 512) * HW + k;
    float s = 0.f;
#pragma unroll 8
    for (int q = 0; q < 512; ++q)
        s += base[(size_t)q * HW];
    partial[((size_t)b * 8 + qc) * HW + k] = s;
}

// ---- kernel 3: reduce 8 partials -> s[b,k] --------------------------------
// grid = (HW/256, NB)
__global__ __launch_bounds__(256)
void colsum_final(const float* __restrict__ partial, float* __restrict__ s) {
    const int b = blockIdx.y;
    const int k = blockIdx.x * 256 + threadIdx.x;
    float t = 0.f;
#pragma unroll
    for (int qc = 0; qc < 8; ++qc)
        t += partial[((size_t)b * 8 + qc) * HW + k];
    s[(size_t)b * HW + k] = t;
}

// ---- kernel 4: out *= 1/(s + 1e-8), float4 grain --------------------------
__global__ __launch_bounds__(256)
void renormalize(float* __restrict__ out, const float* __restrict__ s) {
    size_t idx = (size_t)blockIdx.x * 256 + threadIdx.x;   // float4 index
    size_t e   = idx * 4;                                  // element index
    int b = (int)(e >> 24);                                // / (4096*4096)
    int k = (int)(e & (HW - 1));                           // same row for all 4
    const float* sb = s + (size_t)b * HW + k;
    float4 v = ((float4*)out)[idx];
    v.x *= 1.f / (sb[0] + 1e-8f);
    v.y *= 1.f / (sb[1] + 1e-8f);
    v.z *= 1.f / (sb[2] + 1e-8f);
    v.w *= 1.f / (sb[3] + 1e-8f);
    ((float4*)out)[idx] = v;
}

extern "C" void kernel_launch(void* const* d_in, const int* in_sizes, int n_in,
                              void* d_out, int out_size, void* d_ws, size_t ws_size,
                              hipStream_t stream) {
    const float* fA = (const float*)d_in[0];   // feature_A [8,64,64,256]
    const float* fB = (const float*)d_in[1];   // feature_B [8,64,64,256]
    const float* Wt = (const float*)d_in[2];   // Weight    [64,64,4096] == [q,k]
    float* out = (float*)d_out;                // [8,64,64,4096]

    // workspace: s (NB*HW floats) then partials (NB*8*HW floats) ~= 1.2 MB
    float* s       = (float*)d_ws;
    float* partial = (float*)d_ws + (size_t)NB * HW;

    dim3 g1(HW / 64, HW / 128, NB);            // 64 x 32 x 8 blocks
    corr_gemm_mask<<<g1, 256, 0, stream>>>(fA, fB, Wt, out);

    dim3 g2(HW / 256, 8, NB);                  // 16 x 8 x 8 blocks
    colsum_partial<<<g2, 256, 0, stream>>>(out, partial);

    dim3 g3(HW / 256, NB);                     // 16 x 8 blocks
    colsum_final<<<g3, 256, 0, stream>>>(partial, s);

    const size_t n4 = (size_t)NB * HW * HW / 4;        // 33,554,432 float4s
    renormalize<<<(unsigned)(n4 / 256), 256, 0, stream>>>(out, s);
}
